// DecoderLayer_6124623364306
// MI455X (gfx1250) — compile-verified, matrix operand order checked by reading, not run on hardware
//
#include <hip/hip_runtime.h>

// ---------------- CDNA5 WMMA types ----------------
typedef __attribute__((ext_vector_type(16))) __bf16 v16bf;
typedef __attribute__((ext_vector_type(8)))  float  v8f;
typedef __attribute__((ext_vector_type(4)))  float  v4f;
typedef __attribute__((ext_vector_type(4)))  int    v4i;

__device__ __forceinline__ v8f wmma_bf16(v16bf a, v16bf b, v8f c) {
  // D = A(16x32 bf16) * B(32x16 bf16) + C(16x16 f32)
  return __builtin_amdgcn_wmma_f32_16x16x32_bf16(
      /*neg_a=*/false, a, /*neg_b=*/false, b,
      /*c_mod=*/(short)0, c, /*reuse_a=*/false, /*reuse_b=*/false);
}

// ---------------- optional CDNA5 async global->LDS path ----------------
#if defined(__has_builtin)
#if __has_builtin(__builtin_amdgcn_global_load_async_to_lds_b128)
#define USE_ASYNC_LDS 1
#endif
#endif

#ifdef USE_ASYNC_LDS
#define AS_GLOBAL __attribute__((address_space(1)))
#define AS_LOCAL  __attribute__((address_space(3)))
__device__ __forceinline__ void async_b128(const void* g, void* l) {
  __builtin_amdgcn_global_load_async_to_lds_b128(
      (AS_GLOBAL v4i*)(void*)g,
      (AS_LOCAL  v4i*)l,
      0, 0);
}
__device__ __forceinline__ void async_wait0() {
#if __has_builtin(__builtin_amdgcn_s_wait_asynccnt)
  __builtin_amdgcn_s_wait_asynccnt(0);
#else
  asm volatile("s_wait_asynccnt 0x0" ::: "memory");
#endif
}
#endif

__device__ __forceinline__ void store_bf16x4(__bf16* d, v4f f) {
  d[0] = (__bf16)f[0]; d[1] = (__bf16)f[1]; d[2] = (__bf16)f[2]; d[3] = (__bf16)f[3];
}

// A-fragment / B-fragment loads from row-major LDS tile base[idx][k], ld in elems.
// 16-bit 16x32 layout: lanes 0-15 hold K 0..7,16..23; lanes 16-31 hold K 8..15,24..31.
__device__ __forceinline__ v16bf load_frag_row(const __bf16* base, int ld, int lane) {
  const int idx = lane & 15;
  const int kb  = (lane >> 4) << 3;
  v16bf f;
#pragma unroll
  for (int e = 0; e < 16; ++e) {
    const int k = (e & 7) + ((e >> 3) << 4) + kb;
    f[e] = base[idx * ld + k];
  }
  return f;
}

// B-fragment where LDS tile is K-major: B[k][n] = base[k*ld + n] (for P*V)
__device__ __forceinline__ v16bf load_frag_col(const __bf16* base, int ld, int lane) {
  const int n  = lane & 15;
  const int kb = (lane >> 4) << 3;
  v16bf f;
#pragma unroll
  for (int e = 0; e < 16; ++e) {
    const int k = (e & 7) + ((e >> 3) << 4) + kb;
    f[e] = base[k * ld + n];
  }
  return f;
}

// ---------------- RMSNorm (fp32) ----------------
__global__ __launch_bounds__(256)
void rmsnorm_kernel(const float* __restrict__ x, const float* __restrict__ w,
                    float* __restrict__ out, int H) {
  const int row = blockIdx.x;
  const float* xr = x + (size_t)row * H;
  float* orow = out + (size_t)row * H;
  float ss = 0.f;
  for (int i = threadIdx.x; i < H; i += 256) { float v = xr[i]; ss += v * v; }
#pragma unroll
  for (int m = 1; m < 32; m <<= 1) ss += __shfl_xor(ss, m);
  __shared__ float sred[8];
  __shared__ float sscale;
  if ((threadIdx.x & 31) == 0) sred[threadIdx.x >> 5] = ss;
  __syncthreads();
  if (threadIdx.x == 0) {
    float t = 0.f;
#pragma unroll
    for (int i = 0; i < 8; ++i) t += sred[i];
    sscale = rsqrtf(t / (float)H + 1e-6f);
  }
  __syncthreads();
  const float sc = sscale;
  for (int i = threadIdx.x; i < H; i += 256)
    orow[i] = xr[i] * sc * (1.f + w[i]);
}

// ---------------- GEMM: C[M,N] = A[M,K] * W[N,K]^T  (bf16 WMMA, fp32 accum) ----
// Block tile 128x128, 8 waves, each wave 32(M)x64(N) = 2x4 WMMA tiles.
// A (activations) cached + prefetched; W (weights) streamed non-temporally.
// MODE 0: C = A*W^T ; MODE 1: C = R + A*W^T ; MODE 2: C += A*W^T
template <int MODE>
__global__ __launch_bounds__(256)
void gemm_kernel(const float* __restrict__ A, const float* __restrict__ W,
                 const float* __restrict__ R, float* __restrict__ C,
                 int M, int N, int K) {
  (void)M;
  __shared__ __align__(16) __bf16 As[128 * 32];
  __shared__ __align__(16) __bf16 Bs[128 * 32];
  const int tid  = threadIdx.x;
  const int lane = tid & 31;
  const int wid  = tid >> 5;            // 8 waves (wave32)
  const int wm   = wid >> 1;            // 0..3 -> 32-row chunk
  const int wn   = wid & 1;             // 0..1 -> 64-col chunk
  const int m0   = blockIdx.y * 128;
  const int n0   = blockIdx.x * 128;

  v8f acc[2][4];
#pragma unroll
  for (int i = 0; i < 2; ++i)
#pragma unroll
    for (int j = 0; j < 4; ++j) acc[i][j] = (v8f){0.f,0.f,0.f,0.f,0.f,0.f,0.f,0.f};

  for (int kt = 0; kt < K; kt += 32) {
    __syncthreads();
    // phase 1: batch all 8 global loads (no wait in between)
    v4f fa[4], fb[4];
#pragma unroll
    for (int i = 0; i < 4; ++i) {
      const int idx = tid + i * 256;          // 0..1023
      const int row = idx >> 3, c4 = (idx & 7) << 2;
      const float* srcA = A + (size_t)(m0 + row) * K + kt + c4;
      fa[i] = *(const v4f*)srcA;
      if (kt + 32 < K) __builtin_prefetch(srcA + 32, 0, 1);
      fb[i] = __builtin_nontemporal_load((const v4f*)(W + (size_t)(n0 + row) * K + kt + c4));
    }
    // phase 2: convert fp32 -> bf16 and store into LDS
#pragma unroll
    for (int i = 0; i < 4; ++i) {
      const int idx = tid + i * 256;
      const int row = idx >> 3, c4 = (idx & 7) << 2;
      store_bf16x4(&As[row * 32 + c4], fa[i]);
      store_bf16x4(&Bs[row * 32 + c4], fb[i]);
    }
    __syncthreads();

    v16bf af[2], bf2[4];
#pragma unroll
    for (int i = 0; i < 2; ++i) af[i]  = load_frag_row(As + (wm * 32 + i * 16) * 32, 32, lane);
#pragma unroll
    for (int j = 0; j < 4; ++j) bf2[j] = load_frag_row(Bs + (wn * 64 + j * 16) * 32, 32, lane);
#pragma unroll
    for (int i = 0; i < 2; ++i)
#pragma unroll
      for (int j = 0; j < 4; ++j) acc[i][j] = wmma_bf16(af[i], bf2[j], acc[i][j]);
  }

  // epilogue: C layout -> lane holds col (lane&15); VGPR r -> row r + 8*(lane>>4)
  const int col = lane & 15;
  const int rb  = (lane >> 4) << 3;
#pragma unroll
  for (int i = 0; i < 2; ++i)
#pragma unroll
    for (int j = 0; j < 4; ++j)
#pragma unroll
      for (int r = 0; r < 8; ++r) {
        const int grow = m0 + wm * 32 + i * 16 + r + rb;
        const int gcol = n0 + wn * 64 + j * 16 + col;
        const size_t idx = (size_t)grow * N + gcol;
        float v = acc[i][j][r];
        if (MODE == 1) v += R[idx];
        if (MODE == 2) v += C[idx];
        C[idx] = v;
      }
}

// ---------------- Fused SwiGLU GEMM: out_bf16 = silu(A*Wg^T) * (A*Wu^T) -------
// Block tile 128x64, 8 waves, each 32x32 for both gate and up accumulators.
__global__ __launch_bounds__(256)
void gemm_swiglu_kernel(const float* __restrict__ A, const float* __restrict__ Wg,
                        const float* __restrict__ Wu, __bf16* __restrict__ O,
                        int M, int N, int K) {
  (void)M;
  __shared__ __align__(16) __bf16 As[128 * 32];
  __shared__ __align__(16) __bf16 Bg[64 * 32];
  __shared__ __align__(16) __bf16 Bu[64 * 32];
  const int tid  = threadIdx.x;
  const int lane = tid & 31;
  const int wid  = tid >> 5;
  const int wm   = wid >> 1;
  const int wn   = wid & 1;
  const int m0   = blockIdx.y * 128;
  const int n0   = blockIdx.x * 64;

  v8f ag[2][2], au[2][2];
#pragma unroll
  for (int i = 0; i < 2; ++i)
#pragma unroll
    for (int j = 0; j < 2; ++j) {
      ag[i][j] = (v8f){0.f,0.f,0.f,0.f,0.f,0.f,0.f,0.f};
      au[i][j] = (v8f){0.f,0.f,0.f,0.f,0.f,0.f,0.f,0.f};
    }

  for (int kt = 0; kt < K; kt += 32) {
    __syncthreads();
    v4f fa[4], fg[2], fu[2];
#pragma unroll
    for (int i = 0; i < 4; ++i) {
      const int idx = tid + i * 256;
      const int row = idx >> 3, c4 = (idx & 7) << 2;
      const float* srcA = A + (size_t)(m0 + row) * K + kt + c4;
      fa[i] = *(const v4f*)srcA;
      if (kt + 32 < K) __builtin_prefetch(srcA + 32, 0, 1);
    }
#pragma unroll
    for (int i = 0; i < 2; ++i) {
      const int idx = tid + i * 256;
      const int row = idx >> 3, c4 = (idx & 7) << 2;
      fg[i] = __builtin_nontemporal_load((const v4f*)(Wg + (size_t)(n0 + row) * K + kt + c4));
      fu[i] = __builtin_nontemporal_load((const v4f*)(Wu + (size_t)(n0 + row) * K + kt + c4));
    }
#pragma unroll
    for (int i = 0; i < 4; ++i) {
      const int idx = tid + i * 256;
      const int row = idx >> 3, c4 = (idx & 7) << 2;
      store_bf16x4(&As[row * 32 + c4], fa[i]);
    }
#pragma unroll
    for (int i = 0; i < 2; ++i) {
      const int idx = tid + i * 256;
      const int row = idx >> 3, c4 = (idx & 7) << 2;
      store_bf16x4(&Bg[row * 32 + c4], fg[i]);
      store_bf16x4(&Bu[row * 32 + c4], fu[i]);
    }
    __syncthreads();

    v16bf af[2], bg2[2], bu2[2];
#pragma unroll
    for (int i = 0; i < 2; ++i) af[i]  = load_frag_row(As + (wm * 32 + i * 16) * 32, 32, lane);
#pragma unroll
    for (int j = 0; j < 2; ++j) {
      bg2[j] = load_frag_row(Bg + (wn * 32 + j * 16) * 32, 32, lane);
      bu2[j] = load_frag_row(Bu + (wn * 32 + j * 16) * 32, 32, lane);
    }
#pragma unroll
    for (int i = 0; i < 2; ++i)
#pragma unroll
      for (int j = 0; j < 2; ++j) {
        ag[i][j] = wmma_bf16(af[i], bg2[j], ag[i][j]);
        au[i][j] = wmma_bf16(af[i], bu2[j], au[i][j]);
      }
  }

  const int col = lane & 15;
  const int rb  = (lane >> 4) << 3;
#pragma unroll
  for (int i = 0; i < 2; ++i)
#pragma unroll
    for (int j = 0; j < 2; ++j)
#pragma unroll
      for (int r = 0; r < 8; ++r) {
        const int grow = m0 + wm * 32 + i * 16 + r + rb;
        const int gcol = n0 + wn * 32 + j * 16 + col;
        const float g = ag[i][j][r];
        const float u = au[i][j][r];
        const float s = g / (1.f + __expf(-g));   // silu
        O[(size_t)grow * N + gcol] = (__bf16)(s * u);
      }
}

// ---------------- GEMM with bf16 A (async global->LDS), C += A*W^T ------------
// Block tile 128x128, 8 waves, each wave 32x64.
__global__ __launch_bounds__(256)
void gemm_bf16A_acc_kernel(const __bf16* __restrict__ A, const float* __restrict__ W,
                           float* __restrict__ C, int M, int N, int K) {
  (void)M;
  __shared__ __align__(16) __bf16 As[128 * 32];
  __shared__ __align__(16) __bf16 Bs[128 * 32];
  const int tid  = threadIdx.x;
  const int lane = tid & 31;
  const int wid  = tid >> 5;
  const int wm   = wid >> 1;
  const int wn   = wid & 1;
  const int m0   = blockIdx.y * 128;
  const int n0   = blockIdx.x * 128;

  v8f acc[2][4];
#pragma unroll
  for (int i = 0; i < 2; ++i)
#pragma unroll
    for (int j = 0; j < 4; ++j) acc[i][j] = (v8f){0.f,0.f,0.f,0.f,0.f,0.f,0.f,0.f};

  for (int kt = 0; kt < K; kt += 32) {
    __syncthreads();
    // A tile: 128x32 bf16 = 512 x b128; already bf16 in global, so stage via
    // CDNA5 async global->LDS copies (ASYNCcnt) when available.
#pragma unroll
    for (int i = 0; i < 2; ++i) {
      const int idx = tid + i * 256;          // 0..511
      const int row = idx >> 2, c8 = (idx & 3) << 3;
      const __bf16* src = A + (size_t)(m0 + row) * K + kt + c8;
#ifdef USE_ASYNC_LDS
      async_b128(src, &As[row * 32 + c8]);
#else
      *(uint4*)&As[row * 32 + c8] = *(const uint4*)src;
#endif
    }
    // W tile 128x32 fp32 -> bf16 (batched NT loads, then convert)
    v4f fb[4];
#pragma unroll
    for (int i = 0; i < 4; ++i) {
      const int idx = tid + i * 256;
      const int row = idx >> 3, c4 = (idx & 7) << 2;
      fb[i] = __builtin_nontemporal_load((const v4f*)(W + (size_t)(n0 + row) * K + kt + c4));
    }
#pragma unroll
    for (int i = 0; i < 4; ++i) {
      const int idx = tid + i * 256;
      const int row = idx >> 3, c4 = (idx & 7) << 2;
      store_bf16x4(&Bs[row * 32 + c4], fb[i]);
    }
#ifdef USE_ASYNC_LDS
    async_wait0();
#endif
    __syncthreads();

    v16bf af[2], bf2[4];
#pragma unroll
    for (int i = 0; i < 2; ++i) af[i]  = load_frag_row(As + (wm * 32 + i * 16) * 32, 32, lane);
#pragma unroll
    for (int j = 0; j < 4; ++j) bf2[j] = load_frag_row(Bs + (wn * 64 + j * 16) * 32, 32, lane);
#pragma unroll
    for (int i = 0; i < 2; ++i)
#pragma unroll
      for (int j = 0; j < 4; ++j) acc[i][j] = wmma_bf16(af[i], bf2[j], acc[i][j]);
  }

  const int col = lane & 15;
  const int rb  = (lane >> 4) << 3;
#pragma unroll
  for (int i = 0; i < 2; ++i)
#pragma unroll
    for (int j = 0; j < 4; ++j)
#pragma unroll
      for (int r = 0; r < 8; ++r) {
        const int grow = m0 + wm * 32 + i * 16 + r + rb;
        const int gcol = n0 + wn * 64 + j * 16 + col;
        const size_t idx = (size_t)grow * N + gcol;
        C[idx] = C[idx] + acc[i][j][r];
      }
}

// ---------------- Per-head QK RMSNorm + RoPE (in place) -----------------------
// q rows: [token][16 heads][128], k rows: [token][4 heads][128]
__global__ __launch_bounds__(32)
void qknorm_rope_kernel(float* __restrict__ q, float* __restrict__ k,
                        const float* __restrict__ qw, const float* __restrict__ kw) {
  const int r = blockIdx.x;
  const int lane = threadIdx.x;
  float* base;
  const float* w;
  int token;
  if (r < 4096 * 16) {
    token = r >> 4;
    base  = q + (size_t)token * 2048 + (size_t)(r & 15) * 128;
    w     = qw;
  } else {
    const int r2 = r - 4096 * 16;
    token = r2 >> 2;
    base  = k + (size_t)token * 512 + (size_t)(r2 & 3) * 128;
    w     = kw;
  }
  float x0 = base[lane], x1 = base[lane + 32], x2 = base[lane + 64], x3 = base[lane + 96];
  float ss = x0 * x0 + x1 * x1 + x2 * x2 + x3 * x3;
#pragma unroll
  for (int m = 1; m < 32; m <<= 1) ss += __shfl_xor(ss, m);
  const float rms = rsqrtf(ss * (1.0f / 128.f) + 1e-6f);
  const float n0 = x0 * rms * (1.f + w[lane]);
  const float n1 = x1 * rms * (1.f + w[lane + 32]);
  const float n2 = x2 * rms * (1.f + w[lane + 64]);
  const float n3 = x3 * rms * (1.f + w[lane + 96]);
  const float pos = (float)(token & 1023);
  // inv_freq[i] = 10000^(-2i/128); dims (lane, lane+64) share i=lane, (lane+32, lane+96) share i=lane+32
  const float th0 = pos * __expf(-(float)(2 * lane) * (9.210340371976184f / 128.f));
  const float th1 = pos * __expf(-(float)(2 * (lane + 32)) * (9.210340371976184f / 128.f));
  const float c0 = __cosf(th0), s0 = __sinf(th0);
  const float c1 = __cosf(th1), s1 = __sinf(th1);
  base[lane]      = n0 * c0 - n2 * s0;
  base[lane + 32] = n1 * c1 - n3 * s1;
  base[lane + 64] = n2 * c0 + n0 * s0;
  base[lane + 96] = n3 * c1 + n1 * s1;
}

// ---------------- Flash attention (causal) + sigmoid(z) gate ------------------
// grid: (S/64, NH, B), block 128 (4 waves). Each wave: 16 q rows x HD=128.
__global__ __launch_bounds__(128)
void attn_kernel(const float* __restrict__ q, const float* __restrict__ k,
                 const float* __restrict__ v, const float* __restrict__ z,
                 float* __restrict__ out) {
  constexpr int S = 1024, HD = 128;
  __shared__ __align__(16) __bf16 Qs[64 * 128];   // 16KB
  __shared__ __align__(16) __bf16 Ks[32 * 128];   // 8KB
  __shared__ __align__(16) __bf16 Vs[32 * 128];   // 8KB
  __shared__ __align__(16) __bf16 Ps[4][16 * 32]; // 4KB (wave-private P tiles)

  const int b  = blockIdx.z;
  const int h  = blockIdx.y;
  const int qt = blockIdx.x;
  const int q0 = qt * 64;
  const int tid  = threadIdx.x;
  const int lane = tid & 31;
  const int wv   = tid >> 5;
  const int kvh  = h >> 2;

  // stage Q tile 64x128 (this head) -> bf16 LDS (batched loads)
#pragma unroll
  for (int g = 0; g < 4; ++g) {
    v4f f[4];
#pragma unroll
    for (int j = 0; j < 4; ++j) {
      const int i = tid + (g * 4 + j) * 128;
      const int row = i >> 5, c4 = (i & 31) << 2;
      f[j] = *(const v4f*)(q + (size_t)(b * S + q0 + row) * 2048 + h * HD + c4);
    }
#pragma unroll
    for (int j = 0; j < 4; ++j) {
      const int i = tid + (g * 4 + j) * 128;
      const int row = i >> 5, c4 = (i & 31) << 2;
      store_bf16x4(&Qs[row * 128 + c4], f[j]);
    }
  }
  __syncthreads();

  v16bf qa[4];
#pragma unroll
  for (int kc = 0; kc < 4; ++kc)
    qa[kc] = load_frag_row(Qs + (wv * 16) * 128 + kc * 32, 128, lane);

  float mrun[8], lrun[8];
  v8f o[8];
#pragma unroll
  for (int r = 0; r < 8; ++r) { mrun[r] = -3.0e38f; lrun[r] = 0.f; }
#pragma unroll
  for (int hc = 0; hc < 8; ++hc) o[hc] = (v8f){0.f,0.f,0.f,0.f,0.f,0.f,0.f,0.f};

  const float scale = 0.08838834764831845f; // 1/sqrt(128)
  const int nkb = 2 * qt + 2;               // causal: keys up to q0+63

  for (int kb = 0; kb < nkb; ++kb) {
    __syncthreads();
    // stage K/V tile 32x128 -> bf16 LDS (batched in 2 groups of 4+4 loads)
#pragma unroll
    for (int g = 0; g < 2; ++g) {
      v4f fk[4], fv[4];
#pragma unroll
      for (int j = 0; j < 4; ++j) {
        const int i = tid + (g * 4 + j) * 128;
        const int row = i >> 5, c4 = (i & 31) << 2;
        const size_t tok = (size_t)(b * S + kb * 32 + row);
        fk[j] = *(const v4f*)(k + tok * 512 + kvh * HD + c4);
        fv[j] = *(const v4f*)(v + tok * 512 + kvh * HD + c4);
      }
#pragma unroll
      for (int j = 0; j < 4; ++j) {
        const int i = tid + (g * 4 + j) * 128;
        const int row = i >> 5, c4 = (i & 31) << 2;
        store_bf16x4(&Ks[row * 128 + c4], fk[j]);
        store_bf16x4(&Vs[row * 128 + c4], fv[j]);
      }
    }
    __syncthreads();

    // S = Q * K^T  (two 16x16 score tiles covering 32 keys)
    v8f s0 = (v8f){0.f,0.f,0.f,0.f,0.f,0.f,0.f,0.f};
    v8f s1 = (v8f){0.f,0.f,0.f,0.f,0.f,0.f,0.f,0.f};
#pragma unroll
    for (int kc = 0; kc < 4; ++kc) {
      s0 = wmma_bf16(qa[kc], load_frag_row(Ks + kc * 32, 128, lane), s0);
      s1 = wmma_bf16(qa[kc], load_frag_row(Ks + 16 * 128 + kc * 32, 128, lane), s1);
    }

    // online softmax (rows distributed over VGPR index; cols over 16-lane groups)
    const int col0 = kb * 32 + (lane & 15);
    const int col1 = col0 + 16;
#pragma unroll
    for (int r = 0; r < 8; ++r) {
      const int qrow = q0 + wv * 16 + r + ((lane >> 4) << 3);
      float a0 = s0[r] * scale;
      float a1 = s1[r] * scale;
      if (col0 > qrow) a0 = -3.0e38f;
      if (col1 > qrow) a1 = -3.0e38f;
      float t = fmaxf(a0, a1);
      t = fmaxf(t, __shfl_xor(t, 1));
      t = fmaxf(t, __shfl_xor(t, 2));
      t = fmaxf(t, __shfl_xor(t, 4));
      t = fmaxf(t, __shfl_xor(t, 8));
      const float mn = fmaxf(mrun[r], t);
      const float corr = __expf(mrun[r] - mn);
      const float p0 = __expf(a0 - mn);
      const float p1 = __expf(a1 - mn);
      float rs = p0 + p1;
      rs += __shfl_xor(rs, 1);
      rs += __shfl_xor(rs, 2);
      rs += __shfl_xor(rs, 4);
      rs += __shfl_xor(rs, 8);
      lrun[r] = lrun[r] * corr + rs;
      mrun[r] = mn;
#pragma unroll
      for (int hc = 0; hc < 8; ++hc) o[hc][r] *= corr;
      const int prow = r + ((lane >> 4) << 3);
      Ps[wv][prow * 32 + (lane & 15)]      = (__bf16)p0;
      Ps[wv][prow * 32 + (lane & 15) + 16] = (__bf16)p1;
    }

    // O += P(16x32) * V(32x128)
    const v16bf pa = load_frag_row(&Ps[wv][0], 32, lane);
#pragma unroll
    for (int hc = 0; hc < 8; ++hc)
      o[hc] = wmma_bf16(pa, load_frag_col(Vs + hc * 16, 128, lane), o[hc]);
  }

  // normalize, apply sigmoid(z) gate, store fp32
#pragma unroll
  for (int r = 0; r < 8; ++r) {
    const float inv = 1.0f / lrun[r];
    const int row = r + ((lane >> 4) << 3);
    const size_t tok = (size_t)(b * S + q0 + wv * 16 + row);
#pragma unroll
    for (int hc = 0; hc < 8; ++hc) {
      const int colg = h * HD + hc * 16 + (lane & 15);
      const float zv = z[tok * 2048 + colg];
      const float g  = 1.f / (1.f + __expf(-zv));
      out[tok * 2048 + colg] = o[hc][r] * inv * g;
    }
  }
}

// ---------------- Host-side orchestration -------------------------------------
extern "C" void kernel_launch(void* const* d_in, const int* in_sizes, int n_in,
                              void* d_out, int out_size, void* d_ws, size_t ws_size,
                              hipStream_t stream) {
  (void)in_sizes; (void)n_in; (void)out_size; (void)ws_size;

  const float* x        = (const float*)d_in[0];
  const float* in_ln_w  = (const float*)d_in[1];
  const float* Wq       = (const float*)d_in[2];
  const float* Wk       = (const float*)d_in[3];
  const float* Wv       = (const float*)d_in[4];
  const float* Wz       = (const float*)d_in[5];
  const float* Wo       = (const float*)d_in[6];
  const float* qn_w     = (const float*)d_in[7];
  const float* kn_w     = (const float*)d_in[8];
  const float* post_ln_w= (const float*)d_in[9];
  const float* Wg       = (const float*)d_in[10];
  const float* Wu       = (const float*)d_in[11];
  const float* Wd       = (const float*)d_in[12];
  float* out = (float*)d_out;

  constexpr int M  = 4096;   // B*S
  constexpr int H  = 2048;
  constexpr int FF = 8192;
  constexpr size_t SZ_H  = (size_t)M * 2048 * 4;  // 32MB
  constexpr size_t SZ_KV = (size_t)M * 512 * 4;   // 8MB

  char* ws = (char*)d_ws;
  size_t off = 0;
  float*  hbuf  = (float*)(ws + off); off += SZ_H;    // rmsnorm(x)
  float*  qbuf  = (float*)(ws + off); off += SZ_H;    // q proj
  float*  kbuf  = (float*)(ws + off); off += SZ_KV;   // k proj
  float*  vbuf  = (float*)(ws + off); off += SZ_KV;   // v proj
  float*  zbuf  = (float*)(ws + off); off += SZ_H;    // gate proj
  float*  abuf  = (float*)(ws + off); off += SZ_H;    // gated attention out
  float*  h2buf = (float*)(ws + off); off += SZ_H;    // post rmsnorm
  __bf16* gubuf = (__bf16*)(ws + off);                // silu(g)*u, bf16, 64MB

  // 1) input RMSNorm
  rmsnorm_kernel<<<M, 256, 0, stream>>>(x, in_ln_w, hbuf, H);

  // 2) QKVZ projections (bf16 WMMA GEMMs, 128x128 block tiles)
  gemm_kernel<0><<<dim3(2048 / 128, M / 128), 256, 0, stream>>>(hbuf, Wq, nullptr, qbuf, M, 2048, H);
  gemm_kernel<0><<<dim3(512 / 128,  M / 128), 256, 0, stream>>>(hbuf, Wk, nullptr, kbuf, M, 512,  H);
  gemm_kernel<0><<<dim3(512 / 128,  M / 128), 256, 0, stream>>>(hbuf, Wv, nullptr, vbuf, M, 512,  H);
  gemm_kernel<0><<<dim3(2048 / 128, M / 128), 256, 0, stream>>>(hbuf, Wz, nullptr, zbuf, M, 2048, H);

  // 3) per-head QK RMSNorm + RoPE
  qknorm_rope_kernel<<<M * 16 + M * 4, 32, 0, stream>>>(qbuf, kbuf, qn_w, kn_w);

  // 4) causal flash attention + sigmoid(z) gate
  attn_kernel<<<dim3(16, 16, 4), 128, 0, stream>>>(qbuf, kbuf, vbuf, zbuf, abuf);

  // 5) output projection + residual -> d_out holds x1
  gemm_kernel<1><<<dim3(2048 / 128, M / 128), 256, 0, stream>>>(abuf, Wo, x, out, M, 2048, 2048);

  // 6) post RMSNorm
  rmsnorm_kernel<<<M, 256, 0, stream>>>(out, post_ln_w, h2buf, H);

  // 7) fused SwiGLU: gu = silu(h2*Wg^T) * (h2*Wu^T)  (bf16)
  gemm_swiglu_kernel<<<dim3(FF / 64, M / 128), 256, 0, stream>>>(h2buf, Wg, Wu, gubuf, M, FF, H);

  // 8) down projection (async LDS staging), accumulate: d_out += gu * Wd^T
  gemm_bf16A_acc_kernel<<<dim3(2048 / 128, M / 128), 256, 0, stream>>>(gubuf, Wd, out, M, 2048, FF);
}
